// Mamba_inception_77893526880575
// MI455X (gfx1250) — compile-verified, matrix operand order checked by reading, not run on hardware
//
#include <hip/hip_runtime.h>
#include <hip/hip_bf16.h>

typedef __attribute__((ext_vector_type(16))) _Float16 v16h;
typedef __attribute__((ext_vector_type(8)))  float    v8f;

union AFrag { v16h v; _Float16 h[16]; };
union CFrag { v8f  v; float    f[8];  };
union HQ    { float4 q; _Float16 h[8]; };

#define Bb 8
#define Cc 96
#define Hh 64
#define Ww 64
#define LL 4096
#define DI 192
#define K4 4
#define NSS 16

static __device__ __forceinline__ float siluf(float x) {
  return x / (1.f + __expf(-x));
}
static __device__ __forceinline__ float sigm(float x) {
  return 1.f / (1.f + __expf(-x));
}

// ---------------- weight packing into B-fragment layout -------------------
// dst[((nTile*chunks + chunk)*32 + lane)*16 + e] ; lane: n=lane%16 half=lane/16
// e -> vgpr v=e/2 pos=e%2 ; kk = (v/4)*16 + half*8 + (v%4)*2 + pos
// k >= Kd rows are ZERO-FILLED: consumers may load A unguarded past Kd.
__global__ void k_pack(const float* __restrict__ W, int N, int Kd, int nTiles,
                       int chunks, const float* __restrict__ scale,
                       _Float16* __restrict__ dst) {
  int tid = blockIdx.x * blockDim.x + threadIdx.x;
  int total = nTiles * chunks * 512;
  if (tid >= total) return;
  int e    = tid & 15;
  int lane = (tid >> 4) & 31;
  int rest = tid >> 9;
  int chunk = rest % chunks;
  int nT    = rest / chunks;
  int n = nT * 16 + (lane & 15);
  int half = lane >> 4;
  int v = e >> 1, pos = e & 1;
  int kk = ((v >> 2) << 4) + (half << 3) + ((v & 3) << 1) + pos;
  int k = chunk * 32 + kk;
  float val = 0.f;
  if (n < N && k < Kd) {
    val = W[(size_t)n * Kd + k];
    if (scale) val *= scale[n];
  }
  dst[tid] = (_Float16)val;
}

__global__ void k_bnprep(const float* g, const float* b, const float* m,
                         const float* v, float* scale, float* bias) {
  int i = blockIdx.x * blockDim.x + threadIdx.x;
  if (i >= Cc) return;
  float s = g[i] * rsqrtf(v[i] + 1e-3f);
  scale[i] = s;
  bias[i]  = b[i] - m[i] * s;
}

__global__ void k_aprep(const float* __restrict__ alog, float* __restrict__ an) {
  int i = blockIdx.x * blockDim.x + threadIdx.x;
  if (i >= K4 * DI * NSS) return;
  an[i] = -__expf(alog[i]);
}

// ---------------- generic WMMA GEMM: D(MxN) = A(MxK,f16) * Bpack ----------
// A is f16 row-major; requires lda >= chunks*32 and lda % 8 == 0 (aligned
// unguarded 16B loads; Bpack zero-fills k>=Kd so overread contributes 0).
// mode 0: plain store   D[row*ldd+col]
// mode 1: softplus(x+bias[col])
// mode 2: reverse rows within blocks of p0
// mode 3: NCHW scatter  D[(b*p1+col)*p0 + l], b=row/p0 l=row%p0
// f16out: store _Float16 instead of float
__global__ void k_gemm(const _Float16* __restrict__ A, int lda,
                       const _Float16* __restrict__ Bp, void* __restrict__ Dv,
                       int M, int nTiles, int chunks, int mode,
                       const float* __restrict__ bias, int ldd, int p0, int p1,
                       int f16out) {
  int wid  = (blockIdx.x * blockDim.x + threadIdx.x) >> 5;
  int lane = threadIdx.x & 31;
  int mTiles = M >> 4;
  if (wid >= mTiles * nTiles) return;
  int mT = wid / nTiles, nT = wid % nTiles;
  int half = lane >> 4;
  int mrow = mT * 16 + (lane & 15);
  const _Float16* arow = A + (size_t)mrow * lda + (half << 3);
  const _Float16* brow = Bp + (((size_t)nT * chunks) * 32 + lane) * 16;
  CFrag acc;
#pragma unroll
  for (int i = 0; i < 8; ++i) acc.f[i] = 0.f;
#pragma unroll 2
  for (int ch = 0; ch < chunks; ++ch) {
    AFrag a, bf;
    HQ lo, hi;
    lo.q = *(const float4*)(arow + (ch << 5));
    hi.q = *(const float4*)(arow + (ch << 5) + 16);
#pragma unroll
    for (int e = 0; e < 8; ++e) { a.h[e] = lo.h[e]; a.h[8 + e] = hi.h[e]; }
    bf.v = *(const v16h*)(brow + ((size_t)ch << 9));
    acc.v = __builtin_amdgcn_wmma_f32_16x16x32_f16(false, a.v, false, bf.v,
                                                   (short)0, acc.v, false, false);
  }
  int col = nT * 16 + (lane & 15);
#pragma unroll
  for (int v = 0; v < 8; ++v) {
    int row = mT * 16 + v + (half << 3);
    float val = acc.f[v];
    size_t idx;
    if (mode == 0) {
      idx = (size_t)row * ldd + col;
    } else if (mode == 1) {
      float t = val + bias[col];
      val = (t > 20.f) ? t : log1pf(__expf(t));
      idx = (size_t)row * ldd + col;
    } else if (mode == 2) {
      int bI = row / p0, l = row % p0;
      idx = ((size_t)bI * p0 + (p0 - 1 - l)) * ldd + col;
    } else {
      int bI = row / p0, l = row % p0;
      idx = ((size_t)bI * p1 + col) * p0 + l;
    }
    if (f16out) ((_Float16*)Dv)[idx] = (_Float16)val;
    else        ((float*)Dv)[idx]    = val;
  }
}

// ---------------- 3x3 conv (96->96) implicit GEMM, BN folded, SiLU --------
// LDS holds the im2col tile: 16 pixels x 864 K (f16), row stride 872.
#define KTOT 864
#define KSTR 872
__global__ void k_conv3x3(const float* __restrict__ in,
                          const _Float16* __restrict__ Wp,
                          const float* __restrict__ bias,
                          float* __restrict__ out, int dil) {
  __shared__ __align__(16) _Float16 ldsA[16 * KSTR];
  int tid = threadIdx.x;                 // 192 threads = 6 waves
  int blk = blockIdx.x;                  // b*256 + h*4 + wt
  int wt = blk & 3;
  int h  = (blk >> 2) & 63;
  int b  = blk >> 8;
  int w0 = wt * 16;
  // build im2col tile: element (m, k) ; k = ci*9 + rr*3 + cc
  for (int i = tid; i < 16 * KTOT; i += 192) {
    int m = i & 15, k = i >> 4;
    int ci = k / 9, r = k % 9;
    int rr = r / 3, cc = r % 3;
    int hh = h + (rr - 1) * dil;
    int ww = w0 + m + (cc - 1) * dil;
    float v = 0.f;
    if (hh >= 0 && hh < Hh && ww >= 0 && ww < Ww)
      v = in[(((size_t)b * Cc + ci) * Hh + hh) * Ww + ww];
    ldsA[m * KSTR + k] = (_Float16)v;
  }
  __syncthreads();
  int wave = tid >> 5;                   // cout tile 0..5
  int lane = tid & 31;
  int m = lane & 15, half = lane >> 4;
  const _Float16* arow = ldsA + m * KSTR + (half << 3);
  const _Float16* brow = Wp + (((size_t)wave * 27) * 32 + lane) * 16;
  CFrag acc;
#pragma unroll
  for (int i = 0; i < 8; ++i) acc.f[i] = 0.f;
#pragma unroll 2
  for (int ch = 0; ch < 27; ++ch) {
    AFrag a, bf;
    HQ lo, hi;
    lo.q = *(const float4*)(arow + (ch << 5));
    hi.q = *(const float4*)(arow + (ch << 5) + 16);
#pragma unroll
    for (int e = 0; e < 8; ++e) { a.h[e] = lo.h[e]; a.h[8 + e] = hi.h[e]; }
    bf.v = *(const v16h*)(brow + ((size_t)ch << 9));
    acc.v = __builtin_amdgcn_wmma_f32_16x16x32_f16(false, a.v, false, bf.v,
                                                   (short)0, acc.v, false, false);
  }
  int co = wave * 16 + (lane & 15);
  float bb = bias[co];
#pragma unroll
  for (int v = 0; v < 8; ++v) {
    int w = w0 + v + (half << 3);
    float x = acc.f[v] + bb;
    out[(((size_t)b * Cc + co) * Hh + h) * Ww + w] = siluf(x);
  }
}

// ---------------- LayerNorm over 96 channels of NCHW -> (rows,96) f16 -----
__global__ void k_ln96(const float* __restrict__ x, const float* __restrict__ g,
                       const float* __restrict__ bt, _Float16* __restrict__ outp) {
  int wid  = (blockIdx.x * blockDim.x + threadIdx.x) >> 5;
  int lane = threadIdx.x & 31;
  if (wid >= Bb * LL) return;
  int b = wid >> 12, l = wid & 4095;
  const float* base = x + ((size_t)b * Cc) * LL + l;
  float v0 = base[(size_t)lane * LL];
  float v1 = base[(size_t)(lane + 32) * LL];
  float v2 = base[(size_t)(lane + 64) * LL];
  float s = v0 + v1 + v2, sq = v0 * v0 + v1 * v1 + v2 * v2;
  for (int m = 16; m >= 1; m >>= 1) { s += __shfl_xor(s, m); sq += __shfl_xor(sq, m); }
  float mu = s * (1.f / 96.f);
  float var = sq * (1.f / 96.f) - mu * mu;
  float r = rsqrtf(var + 1e-5f);
  _Float16* orow = outp + (size_t)wid * Cc;
  orow[lane]      = (_Float16)((v0 - mu) * r * g[lane]      + bt[lane]);
  orow[lane + 32] = (_Float16)((v1 - mu) * r * g[lane + 32] + bt[lane + 32]);
  orow[lane + 64] = (_Float16)((v2 - mu) * r * g[lane + 64] + bt[lane + 64]);
}

// ---------------- depthwise 3x3 + SiLU; emit hw and wh layouts (f16) ------
__global__ void k_dwconv(const _Float16* __restrict__ xz, const float* __restrict__ cw,
                         const float* __restrict__ cb, _Float16* __restrict__ xhw,
                         _Float16* __restrict__ xwh) {
  size_t tid = blockIdx.x * (size_t)blockDim.x + threadIdx.x;
  if (tid >= (size_t)Bb * LL * DI) return;
  int d = (int)(tid % DI);
  size_t row = tid / DI;
  int l = (int)(row & 4095), b = (int)(row >> 12);
  int h = l >> 6, w = l & 63;
  float acc = cb[d];
#pragma unroll
  for (int kh = 0; kh < 3; ++kh)
#pragma unroll
    for (int kw = 0; kw < 3; ++kw) {
      int hh = h + kh - 1, ww = w + kw - 1;
      if (hh >= 0 && hh < Hh && ww >= 0 && ww < Ww)
        acc += (float)xz[(((size_t)b * LL + hh * 64 + ww) * 384) + d] *
               cw[d * 9 + kh * 3 + kw];
    }
  float s = siluf(acc);
  xhw[row * DI + d] = (_Float16)s;
  xwh[(((size_t)b * LL) + (w * 64 + h)) * DI + d] = (_Float16)s;
}

// ---------------- selective scan: lane = (b,k,d,n); wave = 2d x 16n -------
__global__ void k_scan(const _Float16* __restrict__ xhw, const _Float16* __restrict__ xwh,
                       const _Float16* __restrict__ xdbl, const _Float16* __restrict__ delta,
                       const float* __restrict__ An, const float* __restrict__ Dsp,
                       float* __restrict__ yc) {
  int tid = blockIdx.x * blockDim.x + threadIdx.x;
  if (tid >= Bb * K4 * DI * NSS) return;
  int n = tid & 15;
  int g = tid >> 4;
  int d = g % DI; g /= DI;
  int k = g & 3;
  int b = g >> 2;
  float A  = An[(k * DI + d) * NSS + n];
  float Dv = Dsp[k * DI + d];
  const _Float16* usrc = ((k & 1) ? xwh : xhw) + ((size_t)b * LL) * DI + d;
  const _Float16* dptr = delta + (((size_t)b * K4 + k) * LL) * DI + d;
  const _Float16* xb   = xdbl  + (((size_t)b * K4 + k) * LL) * 48;
  bool rev = (k >= 2);
  bool tr  = (k & 1);
  float h = 0.f;
  for (int l = 0; l < LL; ++l) {
    int lu = rev ? (LL - 1 - l) : l;
    float u  = (float)usrc[(size_t)lu * DI];
    float dl = (float)dptr[(size_t)l * DI];
    float Bv = (float)xb[(size_t)l * 48 + 6 + n];
    float Cv = (float)xb[(size_t)l * 48 + 22 + n];
    float dA = __expf(dl * A);
    h = dA * h + (dl * u) * Bv;
    float pr = h * Cv;
    pr += __shfl_xor(pr, 1);
    pr += __shfl_xor(pr, 2);
    pr += __shfl_xor(pr, 4);
    pr += __shfl_xor(pr, 8);
    if (n == 0) {
      int ly = tr ? ((lu & 63) * 64 + (lu >> 6)) : lu;
      atomicAdd(&yc[((size_t)b * LL + ly) * DI + d], pr + Dv * u);
    }
  }
}

// ---------------- LN(192) * silu(z) -> f16 gated ---------------------------
__global__ void k_postln(const float* __restrict__ y, const _Float16* __restrict__ xz,
                         const float* __restrict__ g, const float* __restrict__ bt,
                         _Float16* __restrict__ outp) {
  int wid  = (blockIdx.x * blockDim.x + threadIdx.x) >> 5;
  int lane = threadIdx.x & 31;
  if (wid >= Bb * LL) return;
  const float* row = y + (size_t)wid * DI;
  float v[6];
  float s = 0.f, sq = 0.f;
#pragma unroll
  for (int i = 0; i < 6; ++i) {
    v[i] = row[lane + 32 * i];
    s += v[i]; sq += v[i] * v[i];
  }
  for (int m = 16; m >= 1; m >>= 1) { s += __shfl_xor(s, m); sq += __shfl_xor(sq, m); }
  float mu = s * (1.f / 192.f);
  float var = sq * (1.f / 192.f) - mu * mu;
  float r = rsqrtf(var + 1e-5f);
  const _Float16* zrow = xz + (size_t)wid * 384 + DI;
  _Float16* orow = outp + (size_t)wid * DI;
#pragma unroll
  for (int i = 0; i < 6; ++i) {
    int c = lane + 32 * i;
    float ln = (v[i] - mu) * r * g[c] + bt[c];
    orow[c] = (_Float16)(ln * siluf((float)zrow[c]));
  }
}

// ---------------- MLCA ----------------------------------------------------
__global__ void k_mlca_pool(const float* __restrict__ in, float* __restrict__ local) {
  int tid = blockIdx.x * blockDim.x + threadIdx.x;
  if (tid >= Bb * 25 * Cc) return;
  int c  = tid % Cc;
  int pq = (tid / Cc) % 25;
  int b  = tid / (Cc * 25);
  int p = pq / 5, q = pq % 5;
  int rs = (p * 64) / 5, re = ((p + 1) * 64 + 4) / 5;
  int cs = (q * 64) / 5, ce = ((q + 1) * 64 + 4) / 5;
  float s = 0.f;
  for (int r = rs; r < re; ++r)
    for (int cc2 = cs; cc2 < ce; ++cc2)
      s += in[(((size_t)b * Cc + c) * Hh + r) * Ww + cc2];
  local[((size_t)b * 25 + pq) * Cc + c] = s / (float)((re - rs) * (ce - cs));
}

__global__ void k_mlca_local(const float* __restrict__ local,
                             const float* __restrict__ wl,
                             float* __restrict__ attL) {
  int tid = blockIdx.x * blockDim.x + threadIdx.x;
  if (tid >= Bb * 2400) return;
  int j = tid % 2400, b = tid / 2400;
  const float* T = local + (size_t)b * 2400;
  float y = wl[1] * T[j];
  if (j > 0)    y += wl[0] * T[j - 1];
  if (j < 2399) y += wl[2] * T[j + 1];
  attL[tid] = sigm(y);
}

__global__ void k_mlca_global(const float* __restrict__ local,
                              const float* __restrict__ wg,
                              float* __restrict__ attG) {
  int c = blockIdx.x * blockDim.x + threadIdx.x;
  if (c >= Cc) return;
  float sg[8];
  for (int b = 0; b < 8; ++b) {
    float y = 0.f;
    for (int t = 0; t < 3; ++t) {
      int cc2 = c - 1 + t;
      if (cc2 < 0 || cc2 >= Cc) continue;
      float s = 0.f;
      for (int pq = 0; pq < 25; ++pq) s += local[((size_t)b * 25 + pq) * Cc + cc2];
      y += wg[t] * (s * (1.f / 25.f));
    }
    sg[b] = sigm(y);
  }
  for (int p = 0; p < 5; ++p) {
    int s = (p * 8) / 5, e = ((p + 1) * 8 + 4) / 5;
    float a = 0.f;
    for (int b = s; b < e; ++b) a += sg[b];
    attG[c * 5 + p] = a / (float)(e - s);
  }
}

__global__ void k_mlca_apply(const float* __restrict__ in,
                             const float* __restrict__ attL,
                             const float* __restrict__ attG,
                             float* __restrict__ outp,
                             const float* __restrict__ res) {
  size_t tid = blockIdx.x * (size_t)blockDim.x + threadIdx.x;
  if (tid >= (size_t)Bb * Cc * LL) return;
  int hw = (int)(tid & 4095);
  int c  = (int)((tid >> 12) % Cc);
  int b  = (int)(tid / ((size_t)4096 * Cc));
  int h = hw >> 6, w = hw & 63;
  int s1 = (h * 5) >> 6, e1 = ((h + 1) * 5 + 63) >> 6;
  int s2 = (w * 5) >> 6, e2 = ((w + 1) * 5 + 63) >> 6;
  float a = 0.f;
  for (int p = s1; p < e1; ++p)
    for (int q = s2; q < e2; ++q)
      a += 0.5f * attG[c * 5 + p] +
           0.5f * attL[((size_t)b * 25 + p * 5 + q) * Cc + c];
  a /= (float)((e1 - s1) * (e2 - s2));
  size_t iidx = (((size_t)b * Cc + c) << 12) + hw;
  float v = in[iidx] * a + (res ? res[iidx] : 0.f);
  outp[(((size_t)b * 384 + c) << 12) + hw] = v;
}

// ========================== host side =====================================
extern "C" void kernel_launch(void* const* d_in, const int* in_sizes, int n_in,
                              void* d_out, int out_size, void* d_ws, size_t ws_size,
                              hipStream_t stream) {
  const float* X = (const float*)d_in[0];
  const float* cvW[6]; const float* cvG[6]; const float* cvB[6];
  const float* cvM[6]; const float* cvV[6];
  for (int i = 0; i < 6; ++i) {           // cv1,cv3,cv4,cv5,cv6,cv7
    cvW[i] = (const float*)d_in[1 + 5 * i + 0];
    cvG[i] = (const float*)d_in[1 + 5 * i + 1];
    cvB[i] = (const float*)d_in[1 + 5 * i + 2];
    cvM[i] = (const float*)d_in[1 + 5 * i + 3];
    cvV[i] = (const float*)d_in[1 + 5 * i + 4];
  }
  const float* m1wl = (const float*)d_in[31];
  const float* m1wg = (const float*)d_in[32];
  const float* m2wl = (const float*)d_in[33];
  const float* m2wg = (const float*)d_in[34];
  const float* ln1g = (const float*)d_in[35];
  const float* ln1b = (const float*)d_in[36];
  const float* inpW = (const float*)d_in[37];
  const float* dwW  = (const float*)d_in[38];
  const float* dwB  = (const float*)d_in[39];
  const float* xprW = (const float*)d_in[40];
  const float* dtW  = (const float*)d_in[41];
  const float* dtB  = (const float*)d_in[42];
  const float* Alog = (const float*)d_in[43];
  const float* DsP  = (const float*)d_in[44];
  const float* lnG  = (const float*)d_in[45];
  const float* lnB  = (const float*)d_in[46];
  const float* outW = (const float*)d_in[47];
  float* OUT = (float*)d_out;

  size_t off = 0;
  auto alloc = [&](size_t bytes) -> char* {
    off = (off + 255) & ~(size_t)255;
    char* p = (char*)d_ws + off;
    off += bytes;
    return p;
  };
  const size_t M = (size_t)Bb * LL;             // 32768 rows
  _Float16* f_ln    = (_Float16*)alloc(M * 96 * 2);
  _Float16* f_xz    = (_Float16*)alloc(M * 384 * 2);
  _Float16* f_xhw   = (_Float16*)alloc(M * DI * 2);
  _Float16* f_xwh   = (_Float16*)alloc(M * DI * 2);
  _Float16* f_xdbl  = (_Float16*)alloc(M * K4 * 48 * 2);
  _Float16* f_delta = (_Float16*)alloc(M * K4 * DI * 2);
  float* f_yc    = (float*)alloc(M * DI * 4);
  _Float16* f_gate = (_Float16*)alloc(M * DI * 2);
  float* f_sso   = (float*)alloc(M * 96 * 4);
  float* t0      = (float*)alloc(M * 96 * 4);
  float* t1      = (float*)alloc(M * 96 * 4);
  float* f_An    = (float*)alloc(K4 * DI * NSS * 4);
  _Float16* p_cv[6];
  float* s_cv[6]; float* b_cv[6];
  for (int i = 0; i < 6; ++i) {
    p_cv[i] = (_Float16*)alloc((size_t)6 * 27 * 512 * 2);
    s_cv[i] = (float*)alloc(96 * 4);
    b_cv[i] = (float*)alloc(96 * 4);
  }
  _Float16* p_inp = (_Float16*)alloc((size_t)24 * 3 * 512 * 2);
  _Float16* p_out = (_Float16*)alloc((size_t)6 * 6 * 512 * 2);
  _Float16* p_xpr[4]; _Float16* p_dt[4];
  for (int k = 0; k < 4; ++k) {
    p_xpr[k] = (_Float16*)alloc((size_t)3 * 6 * 512 * 2);
    p_dt[k]  = (_Float16*)alloc((size_t)12 * 1 * 512 * 2);
  }
  float* m_local = (float*)alloc((size_t)Bb * 25 * Cc * 4);
  float* m_attL  = (float*)alloc((size_t)Bb * 25 * Cc * 4);
  float* m_attG  = (float*)alloc((size_t)Cc * 5 * 4);

  auto pack = [&](const float* Wsrc, int N, int Kd, int nT, int ch,
                  const float* scale, _Float16* dst) {
    int total = nT * ch * 512;
    k_pack<<<(total + 255) / 256, 256, 0, stream>>>(Wsrc, N, Kd, nT, ch, scale, dst);
  };

  // ---- packing / prep ----
  for (int i = 0; i < 6; ++i) {
    k_bnprep<<<1, 128, 0, stream>>>(cvG[i], cvB[i], cvM[i], cvV[i], s_cv[i], b_cv[i]);
    pack(cvW[i], 96, 864, 6, 27, s_cv[i], p_cv[i]);
  }
  pack(inpW, 384, 96, 24, 3, nullptr, p_inp);
  pack(outW, 96, 192, 6, 6, nullptr, p_out);
  for (int k = 0; k < 4; ++k) {
    pack(xprW + (size_t)k * 38 * 192, 38, 192, 3, 6, nullptr, p_xpr[k]);
    pack(dtW + (size_t)k * 192 * 6, 192, 6, 12, 1, nullptr, p_dt[k]);
  }
  k_aprep<<<(K4 * DI * NSS + 255) / 256, 256, 0, stream>>>(Alog, f_An);

  auto run_mlca = [&](const float* in, const float* wl, const float* wg,
                      float* outBase, const float* res) {
    k_mlca_pool<<<(Bb * 25 * Cc + 255) / 256, 256, 0, stream>>>(in, m_local);
    k_mlca_local<<<(Bb * 2400 + 255) / 256, 256, 0, stream>>>(m_local, wl, m_attL);
    k_mlca_global<<<1, 128, 0, stream>>>(m_local, wg, m_attG);
    k_mlca_apply<<<(int)((M * 96 + 255) / 256), 256, 0, stream>>>(in, m_attL, m_attG,
                                                                  outBase, res);
  };

  // ---- conv / inception branches (WMMA implicit GEMM) ----
  k_conv3x3<<<2048, 192, 0, stream>>>(X, p_cv[0], b_cv[0], t0, 1);      // cv1
  run_mlca(t0, m1wl, m1wg, OUT + (size_t)0 * LL, nullptr);              // y1
  k_conv3x3<<<2048, 192, 0, stream>>>(X, p_cv[3], b_cv[3], t0, 1);      // cv5
  k_conv3x3<<<2048, 192, 0, stream>>>(t0, p_cv[4], b_cv[4], t1, 1);     // cv6
  k_conv3x3<<<2048, 192, 0, stream>>>(t1, p_cv[5], b_cv[5], t0, 3);     // cv7 d=3
  run_mlca(t0, m1wl, m1wg, OUT + (size_t)96 * LL, nullptr);             // y2
  k_conv3x3<<<2048, 192, 0, stream>>>(X, p_cv[1], b_cv[1], t0, 1);      // cv3
  k_conv3x3<<<2048, 192, 0, stream>>>(t0, p_cv[2], b_cv[2], t1, 2);     // cv4 d=2
  run_mlca(t1, m1wl, m1wg, OUT + (size_t)192 * LL, nullptr);            // y3

  // ---- SS2D branch ----
  k_ln96<<<(int)((M * 32 + 255) / 256), 256, 0, stream>>>(X, ln1g, ln1b, f_ln);
  // in_proj: 32768x96 @ 96x384 -> xz (f16)
  k_gemm<<<(2048 * 24 * 32) / 256, 256, 0, stream>>>(f_ln, 96, p_inp, f_xz,
      (int)M, 24, 3, 0, nullptr, 384, 0, 0, 1);
  k_dwconv<<<(int)((M * DI + 255) / 256), 256, 0, stream>>>(f_xz, dwW, dwB, f_xhw, f_xwh);
  // x_proj per (b,k): 4096x192 @ 192x48 (38 valid, reversed store for k>=2)
  for (int b = 0; b < Bb; ++b)
    for (int k = 0; k < 4; ++k) {
      const _Float16* Aa = ((k & 1) ? f_xwh : f_xhw) + (size_t)b * LL * DI;
      _Float16* Dd = f_xdbl + ((size_t)b * 4 + k) * LL * 48;
      k_gemm<<<(256 * 3 * 32) / 256, 256, 0, stream>>>(Aa, DI, p_xpr[k], Dd,
          LL, 3, 6, (k >= 2) ? 2 : 0, nullptr, 48, LL, 0, 1);
    }
  // dt: 4096x6 @ 6x192 + softplus(x + dt_b)   (lda=48 >= 32, Bpack zero past k=6)
  for (int b = 0; b < Bb; ++b)
    for (int k = 0; k < 4; ++k) {
      const _Float16* Aa = f_xdbl + ((size_t)b * 4 + k) * LL * 48;
      _Float16* Dd = f_delta + ((size_t)b * 4 + k) * LL * DI;
      k_gemm<<<(256 * 12 * 32) / 256, 256, 0, stream>>>(Aa, 48, p_dt[k], Dd,
          LL, 12, 1, 1, dtB + (size_t)k * DI, DI, 0, 0, 1);
    }
  hipMemsetAsync(f_yc, 0, M * DI * 4, stream);
  k_scan<<<(Bb * K4 * DI * NSS) / 256, 256, 0, stream>>>(f_xhw, f_xwh, f_xdbl,
      f_delta, f_An, DsP, f_yc);
  k_postln<<<(int)((M * 32 + 255) / 256), 256, 0, stream>>>(f_yc, f_xz, lnG, lnB, f_gate);
  // out_proj: 32768x192 @ 192x96, scatter to NCHW (f32)
  k_gemm<<<(2048 * 6 * 32) / 256, 256, 0, stream>>>(f_gate, DI, p_out, f_sso,
      (int)M, 6, 6, 3, nullptr, 0, LL, 96, 0);
  // x4 = mlca2(ss2d_out) + x
  run_mlca(f_sso, m2wl, m2wg, OUT + (size_t)288 * LL, X);

  (void)in_sizes; (void)n_in; (void)out_size; (void)ws_size;
}
